// EditDistNeuralModel_90460601188382
// MI455X (gfx1250) — compile-verified, matrix operand order checked by reading, not run on hardware
//
#include <hip/hip_runtime.h>
#include <hip/hip_bf16.h>
#include <math.h>

// ---------------------------------------------------------------------------
// EditDistNeuralModel on MI455X (gfx1250): WMMA f16 GEMM for the [65536,64]x
// [64,1089] action-score projection, LDS-staged log-softmax, wavefront DPs.
// Round 3: branch-free WMMA schedule (EXEC all-ones around every v_wmma,
// per ISA 7.12), 4 unrolled dual-accumulator pair iterations + clamped tail.
// ---------------------------------------------------------------------------

typedef __attribute__((ext_vector_type(16))) _Float16 v16h;
typedef __attribute__((ext_vector_type(8)))  float    v8f;

#define T_LEN  256
#define V_LEN  256
#define HID    64
#define NC     1089          // 1 + 32 + 32 + 32*32
#define NTILE  69            // ceil(1089/16)
#define NCP    1104          // NTILE*16 (padded)
#define GRID_CELLS (T_LEN * V_LEN)          // 65536
#define SC_ELEMS ((size_t)GRID_CELLS * NC)  // 71368704

struct BertP {
    const float *word_emb, *pos_emb, *type_emb, *emb_g, *emb_b,
                *Wq, *bq, *Wk, *bk, *Wv, *bv, *Wo, *bo,
                *ln1_g, *ln1_b, *Wi, *bi, *Wf, *bf, *ln2_g, *ln2_b;
};

__device__ __forceinline__ float laex(float a, float b) {
    float m = fmaxf(a, b);
    if (m == -INFINITY) return -INFINITY;
    return m + log1pf(expf(fminf(a, b) - m));
}

// ---------------------------------------------------------------------------
// Kernel 0: pre-swizzle act_W [64][1089] f32 -> f16 B-fragments for
// v_wmma_f32_16x16x32_f16.  Layout: [nt 0..68][kc 0..1][lane 0..31][e 0..15].
// Per lane: col = nt*16 + (lane&15); K(e) = kc*32 + (lane>>4)*8 + (e>>3)*16 + (e&7).
// Also zeroes the entropy accumulator.
// ---------------------------------------------------------------------------
__global__ __launch_bounds__(256) void pack_actW_kernel(const float* __restrict__ actW,
                                                        _Float16* __restrict__ out,
                                                        float* entAcc) {
    int g = blockIdx.x * 256 + threadIdx.x;      // fragment-lane id
    if (g == 0) entAcc[0] = 0.0f;
    if (g >= NTILE * 2 * 32) return;
    int lane = g & 31;
    int kc   = (g >> 5) & 1;
    int nt   = g >> 6;
    int col  = nt * 16 + (lane & 15);
    int kadd = kc * 32 + ((lane >> 4) * 8);
    _Float16 tmp[16];
#pragma unroll
    for (int e = 0; e < 16; ++e) {
        int k = kadd + (e & 7) + ((e >> 3) * 16);
        float x = (col < NC) ? actW[k * NC + col] : 0.0f;
        tmp[e] = (_Float16)x;
    }
#pragma unroll
    for (int e = 0; e < 16; ++e) out[(size_t)g * 16 + e] = tmp[e];
}

// ---------------------------------------------------------------------------
// Kernel 1: single-layer BERT encoder, one block per encoder, one token/thread.
// ---------------------------------------------------------------------------
__global__ __launch_bounds__(256) void bert_kernel(BertP pa, BertP pe,
                                                   const int* __restrict__ ar_ids,
                                                   const int* __restrict__ en_ids,
                                                   float* __restrict__ ar_out,
                                                   float* __restrict__ en_out) {
    const BertP p        = (blockIdx.x == 0) ? pa : pe;
    const int*  ids      = (blockIdx.x == 0) ? ar_ids : en_ids;
    float*      out      = (blockIdx.x == 0) ? ar_out : en_out;

    __shared__ float kS[T_LEN][HID + 1];   // 66.6 KB
    __shared__ float vS[T_LEN][HID + 1];   // 66.6 KB

    const int t = threadIdx.x;
    float h[HID];

    // ---- embedding + layernorm ----
    {
        int wid = ids[t];
        float m = 0.0f;
#pragma unroll 4
        for (int j = 0; j < HID; ++j) {
            float x = p.word_emb[wid * HID + j] + p.pos_emb[t * HID + j] + p.type_emb[j];
            h[j] = x; m += x;
        }
        m *= (1.0f / HID);
        float var = 0.0f;
#pragma unroll 4
        for (int j = 0; j < HID; ++j) { float d = h[j] - m; var += d * d; }
        float inv = rsqrtf(var * (1.0f / HID) + 1e-12f);
#pragma unroll 4
        for (int j = 0; j < HID; ++j) h[j] = (h[j] - m) * inv * p.emb_g[j] + p.emb_b[j];
    }

    // ---- q,k,v ----
    float q[HID];
    for (int j = 0; j < HID; ++j) {
        float aq = p.bq[j], ak = p.bk[j], av = p.bv[j];
#pragma unroll 4
        for (int k = 0; k < HID; ++k) {
            float hk = h[k];
            aq += hk * p.Wq[k * HID + j];
            ak += hk * p.Wk[k * HID + j];
            av += hk * p.Wv[k * HID + j];
        }
        q[j] = aq; kS[t][j] = ak; vS[t][j] = av;
    }
    __syncthreads();

    // ---- attention (4 heads x 16 dims), online max/sum + recompute pass ----
    float ctx[HID];
    for (int hd = 0; hd < 4; ++hd) {
        const int d0 = hd * 16;
        float mx = -INFINITY, sum = 0.0f;
        for (int u = 0; u < T_LEN; ++u) {
            float s = 0.0f;
#pragma unroll
            for (int d = 0; d < 16; ++d) s += q[d0 + d] * kS[u][d0 + d];
            s *= 0.25f;
            if (s > mx) { sum = sum * expf(mx - s) + 1.0f; mx = s; }
            else        { sum += expf(s - mx); }
        }
        float acc[16];
#pragma unroll
        for (int d = 0; d < 16; ++d) acc[d] = 0.0f;
        float inv = 1.0f / sum;
        for (int u = 0; u < T_LEN; ++u) {
            float s = 0.0f;
#pragma unroll
            for (int d = 0; d < 16; ++d) s += q[d0 + d] * kS[u][d0 + d];
            float pr = expf(s * 0.25f - mx) * inv;
#pragma unroll
            for (int d = 0; d < 16; ++d) acc[d] += pr * vS[u][d0 + d];
        }
#pragma unroll
        for (int d = 0; d < 16; ++d) ctx[d0 + d] = acc[d];
    }

    // ---- h = LN(h + ctx @ Wo + bo) ----
    {
        float ho[HID];
        for (int j = 0; j < HID; ++j) {
            float a = p.bo[j];
#pragma unroll 4
            for (int k = 0; k < HID; ++k) a += ctx[k] * p.Wo[k * HID + j];
            ho[j] = h[j] + a;
        }
        float m = 0.0f;
#pragma unroll 4
        for (int j = 0; j < HID; ++j) m += ho[j];
        m *= (1.0f / HID);
        float var = 0.0f;
#pragma unroll 4
        for (int j = 0; j < HID; ++j) { float d = ho[j] - m; var += d * d; }
        float inv = rsqrtf(var * (1.0f / HID) + 1e-12f);
#pragma unroll 4
        for (int j = 0; j < HID; ++j) h[j] = (ho[j] - m) * inv * p.ln1_g[j] + p.ln1_b[j];
    }

    // ---- FFN: gelu(h@Wi+bi)@Wf + bf, residual, LN ----
    {
        float ff[128];
        for (int j = 0; j < 128; ++j) {
            float a = p.bi[j];
#pragma unroll 4
            for (int k = 0; k < HID; ++k) a += h[k] * p.Wi[k * 128 + j];
            ff[j] = 0.5f * a * (1.0f + erff(a * 0.70710678118654752f));
        }
        float ho[HID];
        for (int j = 0; j < HID; ++j) {
            float a = p.bf[j];
#pragma unroll 4
            for (int k = 0; k < 128; ++k) a += ff[k] * p.Wf[k * HID + j];
            ho[j] = h[j] + a;
        }
        float m = 0.0f;
#pragma unroll 4
        for (int j = 0; j < HID; ++j) m += ho[j];
        m *= (1.0f / HID);
        float var = 0.0f;
#pragma unroll 4
        for (int j = 0; j < HID; ++j) { float d = ho[j] - m; var += d * d; }
        float inv = rsqrtf(var * (1.0f / HID) + 1e-12f);
#pragma unroll 4
        for (int j = 0; j < HID; ++j)
            out[t * HID + j] = (ho[j] - m) * inv * p.ln2_g[j] + p.ln2_b[j];
    }
}

// ---------------------------------------------------------------------------
// Kernel 2: per 16 grid cells -> proj+ReLU -> WMMA GEMM (K=64, N=1089) ->
// log_softmax in LDS -> store scores, extract del/ins/sub, entropy.
// ---------------------------------------------------------------------------
__global__ __launch_bounds__(256) void fused_scores_kernel(
        const float* __restrict__ ar_vec, const float* __restrict__ en_vec,
        const float* __restrict__ projW,  const float* __restrict__ projB,
        const v16h*  __restrict__ actW_sw, const float* __restrict__ actB,
        const int* __restrict__ ar_ids,   const int* __restrict__ en_ids,
        float* __restrict__ scores_out,
        float* __restrict__ dels, float* __restrict__ inss, float* __restrict__ subs,
        float* __restrict__ entAcc) {
    __shared__ float    logits[16][NCP];   // 70.6 KB
    __shared__ _Float16 ftA[16][HID];      // 2 KB
    __shared__ float    red[16][17];
    __shared__ float    rowmax[16];
    __shared__ float    rowlse[16];
    __shared__ float    entpart[256];

    const int tid = threadIdx.x;
    const int m0  = blockIdx.x * 16;

    // ---- proj + ReLU -> f16 A tile (each thread: 1 row, 4 cols) ----
    {
        const int r    = tid >> 4;
        const int j0   = (tid & 15) * 4;
        const int cell = m0 + r, t = cell >> 8, v = cell & 255;
        const float* av = ar_vec + t * HID;
        const float* ev = en_vec + v * HID;
        float acc[4];
#pragma unroll
        for (int x = 0; x < 4; ++x) acc[x] = projB[j0 + x];
#pragma unroll 4
        for (int k = 0; k < HID; ++k) {
            float f = av[k];
#pragma unroll
            for (int x = 0; x < 4; ++x) acc[x] += f * projW[k * HID + j0 + x];
        }
#pragma unroll 4
        for (int k = 0; k < HID; ++k) {
            float f = ev[k];
#pragma unroll
            for (int x = 0; x < 4; ++x) acc[x] += f * projW[(HID + k) * HID + j0 + x];
        }
#pragma unroll
        for (int x = 0; x < 4; ++x) ftA[r][j0 + x] = (_Float16)fmaxf(acc[x], 0.0f);
    }
    __syncthreads();

    // ---- WMMA GEMM: branch-free schedule.
    //      Waves take N-tile pairs (nt, nt+8) at stride 16: for NTILE=69 the
    //      first 4 pair-iterations (tiles 0..63) are valid for ALL 8 waves, so
    //      no control flow is needed around the v_wmma chain; the single tail
    //      tile 64+wave is computed by every wave on a clamped index (loads in
    //      bounds, EXEC all-ones) and only the LDS store is predicated.
    {
        const int wave = tid >> 5;
        const int lane = tid & 31;
        v16h a0, a1;
        {
            const int row  = lane & 15;
            const int kadd = (lane >> 4) * 8;
#pragma unroll
            for (int e = 0; e < 16; ++e) {
                int k = kadd + (e & 7) + ((e >> 3) * 16);
                a0[e] = ftA[row][k];
                a1[e] = ftA[row][k + 32];
            }
        }
        const int colL  = lane & 15;
        const int rbase = (lane >> 4) * 8;

#pragma unroll
        for (int i = 0; i < 4; ++i) {
            const int nt0 = wave + i * 16;     // 0..63: always < NTILE
            const int nt1 = nt0 + 8;           // 8..71 capped at 63+8=71? no: wave<8 -> nt1 <= 63+8? max wave=7,i=3 -> nt1=63 < 69
            v16h b00 = actW_sw[(nt0 * 2 + 0) * 32 + lane];
            v16h b01 = actW_sw[(nt0 * 2 + 1) * 32 + lane];
            v16h b10 = actW_sw[(nt1 * 2 + 0) * 32 + lane];
            v16h b11 = actW_sw[(nt1 * 2 + 1) * 32 + lane];
            v8f c0 = {};
            v8f c1 = {};
            c0 = __builtin_amdgcn_wmma_f32_16x16x32_f16(false, a0, false, b00,
                                                        (short)0, c0, false, false);
            c1 = __builtin_amdgcn_wmma_f32_16x16x32_f16(false, a0, false, b10,
                                                        (short)0, c1, false, false);
            c0 = __builtin_amdgcn_wmma_f32_16x16x32_f16(false, a1, false, b01,
                                                        (short)0, c0, false, false);
            c1 = __builtin_amdgcn_wmma_f32_16x16x32_f16(false, a1, false, b11,
                                                        (short)0, c1, false, false);
            {
                const int   col  = nt0 * 16 + colL;   // < 1024 < NC
                const float bias = actB[col];
#pragma unroll
                for (int e = 0; e < 8; ++e) logits[rbase + e][col] = c0[e] + bias;
            }
            {
                const int   col  = nt1 * 16 + colL;   // < 1024+128 <= 1136? no: nt1<=63 -> col<=1023
                const float bias = actB[col];
#pragma unroll
                for (int e = 0; e < 8; ++e) logits[rbase + e][col] = c1[e] + bias;
            }
        }
        // tail: tiles 64..68 belong to waves 0..4; all waves run the WMMA on a
        // clamped tile so EXEC stays all-ones; only stores are predicated.
        {
            const int ntt   = 64 + wave;                    // 64..71
            const int ntc   = (ntt < NTILE) ? ntt : (NTILE - 1);
            v16h b0 = actW_sw[(ntc * 2 + 0) * 32 + lane];
            v16h b1 = actW_sw[(ntc * 2 + 1) * 32 + lane];
            v8f c = {};
            c = __builtin_amdgcn_wmma_f32_16x16x32_f16(false, a0, false, b0,
                                                       (short)0, c, false, false);
            c = __builtin_amdgcn_wmma_f32_16x16x32_f16(false, a1, false, b1,
                                                       (short)0, c, false, false);
            if (ntt < NTILE) {
                const int   col  = ntt * 16 + colL;         // 1024..1103 (NCP padded)
                const float bias = (col < NC) ? actB[col] : 0.0f;
#pragma unroll
                for (int e = 0; e < 8; ++e) logits[rbase + e][col] = c[e] + bias;
            }
        }
    }
    __syncthreads();

    // ---- per-row log-softmax (16 threads per row) ----
    {
        const int r = tid >> 4, s = tid & 15;
        float lm = -INFINITY;
        for (int c = s; c < NC; c += 16) lm = fmaxf(lm, logits[r][c]);
        red[r][s] = lm;
        __syncthreads();
        if (s == 0) {
            float m = red[r][0];
#pragma unroll
            for (int i = 1; i < 16; ++i) m = fmaxf(m, red[r][i]);
            rowmax[r] = m;
        }
        __syncthreads();
        float ls = 0.0f;
        const float rm = rowmax[r];
        for (int c = s; c < NC; c += 16) ls += expf(logits[r][c] - rm);
        red[r][s] = ls;
        __syncthreads();
        if (s == 0) {
            float tot = 0.0f;
#pragma unroll
            for (int i = 0; i < 16; ++i) tot += red[r][i];
            rowlse[r] = rm + logf(tot);
        }
        __syncthreads();
    }

    // ---- store normalized scores (float4, contiguous 16*1089 chunk) ----
    {
        float ent = 0.0f;
        float4* dst = (float4*)(scores_out + (size_t)m0 * NC);  // 69696 B, 16B aligned
        for (int q = tid; q < (16 * NC) / 4; q += 256) {        // 4356 float4s
            const int idx = q * 4;
            float vals[4];
#pragma unroll
            for (int x = 0; x < 4; ++x) {
                const int id = idx + x;
                const int r  = id / NC;
                const int c  = id - r * NC;
                const float s = logits[r][c] - rowlse[r];
                vals[x] = s;
                ent -= s * expf(s);
            }
            float4 o; o.x = vals[0]; o.y = vals[1]; o.z = vals[2]; o.w = vals[3];
            dst[q] = o;
        }
        entpart[tid] = ent;
        __syncthreads();
        if (tid == 0) {
            float tot = 0.0f;
            for (int i = 0; i < 256; ++i) tot += entpart[i];
            atomicAdd(entAcc, tot);
        }
    }

    // ---- gather del / ins / sub scores ----
    if (tid < 16) {
        const int r = tid, cell = m0 + r;
        const int t = cell >> 8, v = cell & 255;
        const int a = ar_ids[t], e = en_ids[v];
        const float lse = rowlse[r];
        dels[cell] = logits[r][1 + a] - lse;
        inss[cell] = logits[r][33 + e] - lse;
        subs[cell] = logits[r][65 + 32 * a + e] - lse;
    }
}

// ---------------------------------------------------------------------------
// Kernel 3a/3b: alpha / beta edit-distance DPs, anti-diagonal wavefront.
// ---------------------------------------------------------------------------
__global__ __launch_bounds__(256) void alpha_kernel(const float* __restrict__ dels,
                                                    const float* __restrict__ inss,
                                                    const float* __restrict__ subs,
                                                    float* __restrict__ alpha) {
    __shared__ float d1[256], d2[256];
    const int v = threadIdx.x;
    for (int d = 0; d < 511; ++d) {
        const int t = d - v;
        const bool act = (t >= 0 && t < 256);
        float nv = 0.0f;
        if (act) {
            if (t == 0 && v == 0)      nv = 0.0f;
            else if (t == 0)           nv = d1[v - 1] + inss[v];
            else if (v == 0)           nv = d1[0] + dels[t * 256];
            else {
                float c = laex(dels[t * 256 + v] + d1[v], subs[t * 256 + v] + d2[v - 1]);
                nv = laex(d1[v - 1] + inss[t * 256 + v], c);
            }
            alpha[t * 256 + v] = nv;
        }
        __syncthreads();
        d2[v] = d1[v];
        __syncthreads();
        if (act) d1[v] = nv;
        __syncthreads();
    }
}

__global__ __launch_bounds__(256) void beta_kernel(const float* __restrict__ dels,
                                                   const float* __restrict__ inss,
                                                   const float* __restrict__ subs,
                                                   float* __restrict__ beta) {
    __shared__ float d1[256], d2[256];
    const int v = threadIdx.x;
    for (int d = 510; d >= 0; --d) {
        const int t = d - v;
        const bool act = (t >= 0 && t < 256);
        float nv = 0.0f;
        if (act) {
            if (t == 255 && v == 255)  nv = 0.0f;
            else if (t == 255)         nv = d1[v + 1] + inss[255 * 256 + v + 1];
            else if (v == 255)         nv = d1[255] + dels[(t + 1) * 256 + 255];
            else {
                float c = laex(dels[(t + 1) * 256 + v] + d1[v],
                               subs[(t + 1) * 256 + v + 1] + d2[v + 1]);
                nv = laex(inss[t * 256 + v + 1] + d1[v + 1], c);
            }
            beta[t * 256 + v] = nv;
        }
        __syncthreads();
        d2[v] = d1[v];
        __syncthreads();
        if (act) d1[v] = nv;
        __syncthreads();
    }
}

// ---------------------------------------------------------------------------
// Kernel 4a: zero exp(ec); 4b: 3-class scatter per cell + final scalars.
// ---------------------------------------------------------------------------
__global__ void zero_ec_kernel(float4* __restrict__ ec4, long long n4) {
    long long i = (long long)blockIdx.x * blockDim.x + threadIdx.x;
    long long stride = (long long)gridDim.x * blockDim.x;
    float4 z; z.x = 0.f; z.y = 0.f; z.z = 0.f; z.w = 0.f;
    for (; i < n4; i += stride) ec4[i] = z;
}

__global__ __launch_bounds__(256) void ec_scatter_kernel(
        const float* __restrict__ alpha, const float* __restrict__ beta,
        const float* __restrict__ dels,  const float* __restrict__ inss,
        const float* __restrict__ subs,
        const int* __restrict__ ar_ids,  const int* __restrict__ en_ids,
        const float* __restrict__ entAcc,
        float* __restrict__ ec, float* __restrict__ outEnt, float* __restrict__ outAlpha) {
    const int cell = blockIdx.x * 256 + threadIdx.x;
    const int t = cell >> 8, v = cell & 255;
    const float b = beta[cell];
    float dv = (t >= 1) ? alpha[(t - 1) * 256 + v] + dels[cell] + b : -INFINITY;
    float iv = (v >= 1) ? alpha[t * 256 + (v - 1)] + inss[cell] + b : -INFINITY;
    float sv = (t >= 1 && v >= 1) ? alpha[(t - 1) * 256 + (v - 1)] + subs[cell] + b : -INFINITY;
    float L = laex(laex(dv, iv), sv);
    if (L != -INFINITY) {
        const int a = ar_ids[t], e = en_ids[v];
        const size_t base = (size_t)cell * NC;
        ec[base + 1 + a]           = expf(dv - L);
        ec[base + 33 + e]          = expf(iv - L);
        ec[base + 65 + 32 * a + e] = expf(sv - L);
    }
    if (cell == 0) {
        outEnt[0]   = entAcc[0];
        outAlpha[0] = alpha[255 * 256 + 255];
    }
}

// ---------------------------------------------------------------------------
// Host launcher
// ---------------------------------------------------------------------------
static BertP make_bertp(void* const* d_in, int base) {
    BertP p;
    const float** f = (const float**)&p;
    for (int i = 0; i < 21; ++i) f[i] = (const float*)d_in[base + i];
    return p;
}

extern "C" void kernel_launch(void* const* d_in, const int* in_sizes, int n_in,
                              void* d_out, int out_size, void* d_ws, size_t ws_size,
                              hipStream_t stream) {
    (void)in_sizes; (void)n_in; (void)out_size; (void)ws_size;

    const int*   ar_ids = (const int*)d_in[0];
    const int*   en_ids = (const int*)d_in[1];
    BertP pa = make_bertp(d_in, 2);
    BertP pe = make_bertp(d_in, 23);
    const float* projW = (const float*)d_in[44];
    const float* projB = (const float*)d_in[45];
    const float* actW  = (const float*)d_in[46];
    const float* actB  = (const float*)d_in[47];

    // workspace layout
    char* base = (char*)d_ws;
    float*    ar_vec = (float*)(base);                 //  65536 B
    float*    en_vec = (float*)(base + 65536);         //  65536 B
    _Float16* actw   = (_Float16*)(base + 131072);     // 141312 B (32B aligned)
    float*    dels   = (float*)(base + 272384);        // 262144 B
    float*    inss   = (float*)(base + 534528);
    float*    subs   = (float*)(base + 796672);
    float*    alpha  = (float*)(base + 1058816);
    float*    beta   = (float*)(base + 1320960);
    float*    entAcc = (float*)(base + 1583104);

    float* out     = (float*)d_out;
    float* scores  = out;                  // [65536][1089]
    float* ec      = out + SC_ELEMS;       // [65536][1089]
    float* outEnt  = out + 2 * SC_ELEMS;
    float* outAlp  = outEnt + 1;

    pack_actW_kernel<<<(NTILE * 2 * 32 + 255) / 256, 256, 0, stream>>>(actW, actw, entAcc);
    bert_kernel<<<2, 256, 0, stream>>>(pa, pe, ar_ids, en_ids, ar_vec, en_vec);
    fused_scores_kernel<<<GRID_CELLS / 16, 256, 0, stream>>>(
        ar_vec, en_vec, projW, projB, (const v16h*)actw, actB,
        ar_ids, en_ids, scores, dels, inss, subs, entAcc);
    alpha_kernel<<<1, 256, 0, stream>>>(dels, inss, subs, alpha);
    beta_kernel<<<1, 256, 0, stream>>>(dels, inss, subs, beta);
    zero_ec_kernel<<<8192, 256, 0, stream>>>((float4*)ec, (long long)(SC_ELEMS / 4));
    ec_scatter_kernel<<<GRID_CELLS / 256, 256, 0, stream>>>(
        alpha, beta, dels, inss, subs, ar_ids, en_ids, entAcc, ec, outEnt, outAlp);
}
